// EmbedMatcher_35150012350853
// MI455X (gfx1250) — matrix operand order repaired and checked by reading, not compiled, not run on hardware
//
#include <hip/hip_runtime.h>
#include <hip/hip_bf16.h>
#include <math.h>

#define KNB      50
#define PAD_IDX  200000
#define NEG_BIG  (-1.0e9f)

typedef __attribute__((ext_vector_type(8)))  float  v8f;
typedef __attribute__((ext_vector_type(16))) __bf16 v16bf;

union FragU { uint4 q[2]; v16bf v; };

__device__ __forceinline__ unsigned short f2bf(float f) {
  union { float f; unsigned int u; } c; c.f = f;
  unsigned int u = c.u;
  unsigned int r = u + 0x7FFFu + ((u >> 16) & 1u);
  return (unsigned short)(r >> 16);
}

__device__ __forceinline__ v8f wmma_bf(v16bf a, v16bf b, v8f c) {
  return __builtin_amdgcn_wmma_f32_16x16x32_bf16(false, a, false, b, (short)0, c, false, false);
}

__device__ __forceinline__ float sigf(float x) { return 1.0f / (1.0f + expf(-x)); }

// ---------------------------------------------------------------------------
// Pack W (row-major [N, ld], using first Kd cols) into WMMA B-fragment layout:
// frag = kt*(N/16)+nt ; per lane (col = nt*16 + lane%16, kbase = kt*32 + (lane/16)*8)
// 16 bf16 per lane: k = kbase + e (e<8), kbase + 16 + (e-8) (e>=8).
// ---------------------------------------------------------------------------
__global__ __launch_bounds__(256) void pack_b_kernel(const float* __restrict__ W,
                                                     unsigned short* __restrict__ Out,
                                                     int N, int Kd, int ld) {
  int idx = blockIdx.x * blockDim.x + threadIdx.x;
  int total = (Kd >> 5) * (N >> 4) * 32;
  if (idx >= total) return;
  int lane = idx & 31;
  int frag = idx >> 5;
  int NT = N >> 4;
  int kt = frag / NT, nt = frag % NT;
  int col = nt * 16 + (lane & 15);
  int kb  = kt * 32 + ((lane >> 4) << 3);
  unsigned short* dst = Out + ((size_t)frag * 32 + lane) * 16;
  const float* src = W + (size_t)col * ld + kb;
#pragma unroll
  for (int e = 0; e < 16; ++e) dst[e] = f2bf(src[e + (e & 8)]);
}

// zero pad rows [5,16) of support vectors
__global__ __launch_bounds__(256) void zpad_kernel(float* sv, unsigned short* svbf) {
  for (int i = threadIdx.x; i < 11 * 256; i += 256) {
    int r = 5 + i / 256, c = i % 256;
    sv[r * 256 + c] = 0.0f;
    svbf[r * 256 + c] = 0;
  }
}

// ---------------------------------------------------------------------------
// Fused GCN neighbor encoder: one workgroup (128 thr = 4 waves) per node-side.
// Dynamic LDS layout (padded strides for bank-conflict-free access):
//   Abuf : u16 [64][264]  (A tile, bf16)       33792 B
//   proj : f32 [64][129]                       33024 B
//   selfe/scores/wts/aggs/h1b/relpad/misc      ~2 KB
// ---------------------------------------------------------------------------
#define NB_AS       264
#define NB_PS       129
#define OFF_PROJ    (64 * NB_AS * 2)            /* 33792 */
#define OFF_SELFE   (OFF_PROJ + 64 * NB_PS * 4) /* 66816 */
#define OFF_SCORES  (OFF_SELFE + 512)
#define OFF_WTS     (OFF_SCORES + 256)
#define OFF_AGGS    (OFF_WTS + 256)
#define OFF_H1B     (OFF_AGGS + 512)
#define OFF_RELPAD  (OFF_H1B + 256)
#define OFF_MISC    (OFF_RELPAD + 256)
#define NB_SMEM     (OFF_MISC + 16)

__global__ __launch_bounds__(128) void neigh_kernel(
    const int* __restrict__ conns, const int* __restrict__ deg,
    const int* __restrict__ selfp,              // element at n*2
    const float* __restrict__ emb, const unsigned short* __restrict__ WpGcn,
    const float* __restrict__ lin_b, const float* __restrict__ gcn_b,
    const float* __restrict__ temp_p,
    const float* __restrict__ cgW1, const float* __restrict__ cgb1,
    const float* __restrict__ cglng, const float* __restrict__ cglnb,
    const float* __restrict__ cgW2, const float* __restrict__ cgb2,
    float* __restrict__ outv, unsigned short* __restrict__ outbf, int out_off) {
  extern __shared__ __align__(16) char smem[];
  unsigned short (*Abuf)[NB_AS] = (unsigned short(*)[NB_AS])smem;
  float (*proj)[NB_PS] = (float(*)[NB_PS])(smem + OFF_PROJ);
  float* selfe  = (float*)(smem + OFF_SELFE);
  float* scores = (float*)(smem + OFF_SCORES);
  float* wts    = (float*)(smem + OFF_WTS);
  float* aggs   = (float*)(smem + OFF_AGGS);
  float* h1b    = (float*)(smem + OFF_H1B);
  int*   relpad = (int*)(smem + OFF_RELPAD);
  float* misc   = (float*)(smem + OFF_MISC);   // [0]=mean [1]=rstd [2]=gate

  const int n = blockIdx.x;
  const int tid = threadIdx.x;

  // --- build A tile (rows: neighbors, cols: [emb[rel], emb[ent]]) ---
  {
    int r = tid >> 1, half = tid & 1;
    int src = PAD_IDX;
    if (r < KNB) src = conns[((size_t)n * KNB + r) * 2 + half];
    if (half == 0) relpad[r] = (r < KNB) ? (src == PAD_IDX) : 1;
    int base = half * 128;
    if (r < KNB) {
      const float* erow = emb + (size_t)src * 128;
#pragma unroll 4
      for (int c = 0; c < 128; c += 4) {
        float4 f = *(const float4*)(erow + c);
        Abuf[r][base + c + 0] = f2bf(f.x);
        Abuf[r][base + c + 1] = f2bf(f.y);
        Abuf[r][base + c + 2] = f2bf(f.z);
        Abuf[r][base + c + 3] = f2bf(f.w);
      }
    } else {
      for (int c = 0; c < 128; ++c) Abuf[r][base + c] = 0;
    }
    int sid = selfp[(size_t)n * 2];
    selfe[tid] = emb[(size_t)sid * 128 + tid];
  }
  __syncthreads();

  // --- WMMA GEMM: proj[64,128] = leakyrelu(A[64,256] x WpGcn + bias) ---
  {
    int wave = tid >> 5, lane = tid & 31;
    int m0 = wave * 16;
    int rr = m0 + (lane & 15);
    int ks = (lane >> 4) << 3;
    FragU af[8];
#pragma unroll
    for (int kt = 0; kt < 8; ++kt) {
      af[kt].q[0] = *(const uint4*)&Abuf[rr][kt * 32 + ks];
      af[kt].q[1] = *(const uint4*)&Abuf[rr][kt * 32 + 16 + ks];
    }
#pragma unroll
    for (int nt = 0; nt < 8; ++nt) {
      v8f acc = {0.f, 0.f, 0.f, 0.f, 0.f, 0.f, 0.f, 0.f};
#pragma unroll
      for (int kt = 0; kt < 8; ++kt) {
        FragU b;
        const unsigned short* bp = WpGcn + ((size_t)(kt * 8 + nt) * 32 + lane) * 16;
        b.q[0] = *(const uint4*)bp;
        b.q[1] = *(const uint4*)(bp + 8);
        acc = wmma_bf(af[kt].v, b.v, acc);
      }
      int col = nt * 16 + (lane & 15);
      float bias = lin_b[col] + gcn_b[col];
#pragma unroll
      for (int v = 0; v < 8; ++v) {
        int row = m0 + v + ((lane >> 4) << 3);
        float x = acc[v] + bias;
        proj[row][col] = x > 0.f ? x : 0.01f * x;   // rows >= KNB never read
      }
    }
  }
  __syncthreads();

  // --- scores + mask ---
  if (tid < 64) {
    float s = NEG_BIG;
    if (tid < KNB) {
      float acc = 0.f;
      for (int j = 0; j < 128; ++j) acc += proj[tid][j] * selfe[j];
      s = relpad[tid] ? NEG_BIG : acc;
    }
    scores[tid] = s;
  }
  __syncthreads();
  if (tid == 0) {
    float m = scores[0];
    for (int k = 1; k < KNB; ++k) m = fmaxf(m, scores[k]);
    float ssum = 0.f;
    for (int k = 0; k < KNB; ++k) { float e = expf(scores[k] - m); wts[k] = e; ssum += e; }
    float inv = 1.0f / ssum;
    for (int k = 0; k < KNB; ++k) wts[k] *= inv;
  }
  __syncthreads();

  // --- agg ---
  {
    float a = 0.f;
    for (int k = 0; k < KNB; ++k) a += wts[k] * proj[k][tid];
    aggs[tid] = a;
  }
  __syncthreads();

  // --- gate MLP: relu(LN(agg@W1^T+b1)) -> logit -> sigmoid ---
  if (tid < 64) {
    float u = cgb1[tid];
    const float* wrow = cgW1 + (size_t)tid * 128;
    for (int j = 0; j < 128; ++j) u += aggs[j] * wrow[j];
    h1b[tid] = u;
  }
  __syncthreads();
  if (tid == 0) {
    float m = 0.f;
    for (int j = 0; j < 64; ++j) m += h1b[j];
    m *= (1.0f / 64.0f);
    float v = 0.f;
    for (int j = 0; j < 64; ++j) { float d = h1b[j] - m; v += d * d; }
    v *= (1.0f / 64.0f);
    misc[0] = m; misc[1] = rsqrtf(v + 1e-5f);
  }
  __syncthreads();
  if (tid < 64) {
    float z = (h1b[tid] - misc[0]) * misc[1] * cglng[tid] + cglnb[tid];
    h1b[tid] = fmaxf(z, 0.f);
  }
  __syncthreads();
  if (tid == 0) {
    float logit = cgb2[0];
    for (int j = 0; j < 64; ++j) logit += h1b[j] * cgW2[j];
    float t = fminf(fmaxf(temp_p[0], 0.1f), 5.0f);
    float g = sigf(logit / t);
    if (deg[n] <= 0) g = 0.f;
    misc[2] = g;
  }
  __syncthreads();
  {
    float o = tanhf(selfe[tid] + misc[2] * aggs[tid]);
    size_t oi = (size_t)n * 256 + out_off + tid;
    outv[oi] = o;
    outbf[oi] = f2bf(o);
  }
}

// ---------------------------------------------------------------------------
// GEMM1 of support_enc: Out_bf[M,512] = relu(A_bf[M,256] x Wp + b1)
// Tile: 16 rows x 512 cols per WG (256 thr = 8 waves, 4 ntiles each).
// ---------------------------------------------------------------------------
__global__ __launch_bounds__(256) void gemm_relu512_kernel(
    const unsigned short* __restrict__ A, const unsigned short* __restrict__ Bp,
    const float* __restrict__ bias, unsigned short* __restrict__ Out) {
  int m0 = blockIdx.x * 16;
  int wave = threadIdx.x >> 5, lane = threadIdx.x & 31;
  size_t rowoff = (size_t)(m0 + (lane & 15)) * 256;
  int ks = (lane >> 4) << 3;
  FragU af[8];
#pragma unroll
  for (int kt = 0; kt < 8; ++kt) {
    af[kt].q[0] = *(const uint4*)(A + rowoff + kt * 32 + ks);
    af[kt].q[1] = *(const uint4*)(A + rowoff + kt * 32 + 16 + ks);
  }
#pragma unroll
  for (int nti = 0; nti < 4; ++nti) {
    int nt = wave * 4 + nti;
    v8f acc = {0.f, 0.f, 0.f, 0.f, 0.f, 0.f, 0.f, 0.f};
#pragma unroll
    for (int kt = 0; kt < 8; ++kt) {
      FragU b;
      const unsigned short* bp = Bp + ((size_t)(kt * 32 + nt) * 32 + lane) * 16;
      b.q[0] = *(const uint4*)bp;
      b.q[1] = *(const uint4*)(bp + 8);
      acc = wmma_bf(af[kt].v, b.v, acc);
    }
    int col = nt * 16 + (lane & 15);
    float bv = bias[col];
#pragma unroll
    for (int v = 0; v < 8; ++v) {
      int row = m0 + v + ((lane >> 4) << 3);
      float x = acc[v] + bv;
      Out[(size_t)row * 512 + col] = f2bf(fmaxf(x, 0.f));
    }
  }
}

// ---------------------------------------------------------------------------
// GEMM2 of support_enc + residual + LayerNorm:
// Y[M,256] = LN(A_bf[M,512] x Wp + b2 + X) ; outputs f32 + bf16.
// Tile: 16 rows x 256 cols per WG (8 waves, 2 ntiles each).
// ---------------------------------------------------------------------------
__global__ __launch_bounds__(256) void gemm_ln256_kernel(
    const unsigned short* __restrict__ A, const unsigned short* __restrict__ Bp,
    const float* __restrict__ b2, const float* __restrict__ X,
    const float* __restrict__ lng, const float* __restrict__ lnb,
    float* __restrict__ Outf, unsigned short* __restrict__ Outbf) {
  __shared__ float Ct[16][257];
  __shared__ float mu[16], rs[16];
  int m0 = blockIdx.x * 16;
  int wave = threadIdx.x >> 5, lane = threadIdx.x & 31;
  size_t rowoff = (size_t)(m0 + (lane & 15)) * 512;
  int ks = (lane >> 4) << 3;
#pragma unroll
  for (int nti = 0; nti < 2; ++nti) {
    int nt = wave * 2 + nti;
    v8f acc = {0.f, 0.f, 0.f, 0.f, 0.f, 0.f, 0.f, 0.f};
#pragma unroll
    for (int kt = 0; kt < 16; ++kt) {
      FragU a, b;
      a.q[0] = *(const uint4*)(A + rowoff + kt * 32 + ks);
      a.q[1] = *(const uint4*)(A + rowoff + kt * 32 + 16 + ks);
      const unsigned short* bp = Bp + ((size_t)(kt * 16 + nt) * 32 + lane) * 16;
      b.q[0] = *(const uint4*)bp;
      b.q[1] = *(const uint4*)(bp + 8);
      acc = wmma_bf(a.v, b.v, acc);
    }
    int col = nt * 16 + (lane & 15);
    float bv = b2[col];
#pragma unroll
    for (int v = 0; v < 8; ++v) {
      int r16 = v + ((lane >> 4) << 3);
      Ct[r16][col] = acc[v] + bv + X[(size_t)(m0 + r16) * 256 + col];
    }
  }
  __syncthreads();
  if (threadIdx.x < 16) {
    int r = threadIdx.x;
    float m = 0.f;
    for (int c = 0; c < 256; ++c) m += Ct[r][c];
    m *= (1.0f / 256.0f);
    float v = 0.f;
    for (int c = 0; c < 256; ++c) { float d = Ct[r][c] - m; v += d * d; }
    v *= (1.0f / 256.0f);
    mu[r] = m; rs[r] = rsqrtf(v + 1e-5f);
  }
  __syncthreads();
  {
    int c = threadIdx.x;
    float g = lng[c], bb = lnb[c];
#pragma unroll
    for (int r = 0; r < 16; ++r) {
      float val = (Ct[r][c] - mu[r]) * rs[r] * g + bb;
      size_t oi = (size_t)(m0 + r) * 256 + c;
      Outf[oi] = val;
      Outbf[oi] = f2bf(val);
    }
  }
}

// ---------------------------------------------------------------------------
// LSTM gate GEMM: Out[M,2048] = A_bf[M,256] x Wp (+b1) (+b2) (+AddFull)
// Tile: 64 rows x 128 cols per WG (128 thr = 4 waves = row tiles).
// ---------------------------------------------------------------------------
__global__ __launch_bounds__(128) void gemm_nt2048_kernel(
    const unsigned short* __restrict__ A, const unsigned short* __restrict__ Bp,
    const float* __restrict__ addF, const float* __restrict__ b1,
    const float* __restrict__ b2, float* __restrict__ Out) {
  int wave = threadIdx.x >> 5, lane = threadIdx.x & 31;
  int m0 = blockIdx.x * 64 + wave * 16;
  int nbase = blockIdx.y * 8;
  size_t rowoff = (size_t)(m0 + (lane & 15)) * 256;
  int ks = (lane >> 4) << 3;
  FragU af[8];
#pragma unroll
  for (int kt = 0; kt < 8; ++kt) {
    af[kt].q[0] = *(const uint4*)(A + rowoff + kt * 32 + ks);
    af[kt].q[1] = *(const uint4*)(A + rowoff + kt * 32 + 16 + ks);
  }
#pragma unroll
  for (int nti = 0; nti < 8; ++nti) {
    int nt = nbase + nti;
    v8f acc = {0.f, 0.f, 0.f, 0.f, 0.f, 0.f, 0.f, 0.f};
#pragma unroll
    for (int kt = 0; kt < 8; ++kt) {
      FragU b;
      const unsigned short* bp = Bp + ((size_t)(kt * 128 + nt) * 32 + lane) * 16;
      b.q[0] = *(const uint4*)bp;
      b.q[1] = *(const uint4*)(bp + 8);
      if (kt < 7)
        __builtin_prefetch(Bp + ((size_t)((kt + 1) * 128 + nt) * 32 + lane) * 16, 0, 1);
      acc = wmma_bf(af[kt].v, b.v, acc);
    }
    int col = nt * 16 + (lane & 15);
    float bb = (b1 ? b1[col] : 0.f) + (b2 ? b2[col] : 0.f);
#pragma unroll
    for (int v = 0; v < 8; ++v) {
      int row = m0 + v + ((lane >> 4) << 3);
      size_t oi = (size_t)row * 2048 + col;
      float z = acc[v] + bb;
      if (addF) z += addF[oi];
      Out[oi] = z;
    }
  }
}

// support_g = mean over 5 rows of s_enc
__global__ __launch_bounds__(256) void support_mean_kernel(const float* __restrict__ senc,
                                                           float* __restrict__ sg) {
  int j = threadIdx.x;
  float a = 0.f;
  for (int r = 0; r < 5; ++r) a += senc[r * 256 + j];
  sg[j] = a * 0.2f;
}

// sgterm[n] = sum_k sg[k] * Whh[n,256+k]
__global__ __launch_bounds__(256) void sgterm_kernel(const float* __restrict__ Whh,
                                                     const float* __restrict__ sg,
                                                     float* __restrict__ sgterm) {
  int n = blockIdx.x * blockDim.x + threadIdx.x;
  if (n >= 2048) return;
  const float* wr = Whh + (size_t)n * 512 + 256;
  float a = 0.f;
  for (int k = 0; k < 256; ++k) a += sg[k] * wr[k];
  sgterm[n] = a;
}

// LSTM cell elementwise
__global__ __launch_bounds__(256) void lstm_cell_kernel(
    const float* __restrict__ gates, float* __restrict__ c,
    const float* __restrict__ qe, float* __restrict__ h,
    unsigned short* __restrict__ hbf, int step) {
  int idx = blockIdx.x * blockDim.x + threadIdx.x;
  int n = idx >> 9, j = idx & 511;
  size_t base = (size_t)n * 2048;
  float gi = gates[base + j];
  float gf = gates[base + 512 + j];
  float gg = gates[base + 1024 + j];
  float go = gates[base + 1536 + j];
  float cold = step ? c[(size_t)n * 512 + j] : 0.f;
  float cn = sigf(gf) * cold + sigf(gi) * tanhf(gg);
  c[(size_t)n * 512 + j] = cn;
  if (j < 256) {
    float hv = qe[(size_t)n * 256 + j] + sigf(go) * tanhf(cn);
    h[(size_t)n * 256 + j] = hv;
    hbf[(size_t)n * 256 + j] = f2bf(hv);
  }
}

// out[n] = dot(h[n,:], support_g)
__global__ __launch_bounds__(128) void final_dot_kernel(const float* __restrict__ h,
                                                        const float* __restrict__ sg,
                                                        float* __restrict__ out) {
  __shared__ float red[128];
  int n = blockIdx.x, t = threadIdx.x;
  const float* hr = h + (size_t)n * 256;
  red[t] = hr[t] * sg[t] + hr[128 + t] * sg[128 + t];
  __syncthreads();
  for (int s = 64; s > 0; s >>= 1) {
    if (t < s) red[t] += red[t + s];
    __syncthreads();
  }
  if (t == 0) out[n] = red[0];
}

extern "C" void kernel_launch(void* const* d_in, const int* in_sizes, int n_in,
                              void* d_out, int out_size, void* d_ws, size_t ws_size,
                              hipStream_t stream) {
  (void)in_sizes; (void)n_in; (void)out_size; (void)ws_size;
  const int*   query     = (const int*)d_in[0];
  const int*   support   = (const int*)d_in[1];
  const int*   q_l1      = (const int*)d_in[2];
  const int*   q_deg_l   = (const int*)d_in[3];
  const int*   q_r1      = (const int*)d_in[4];
  const int*   q_deg_r   = (const int*)d_in[5];
  const int*   s_l1      = (const int*)d_in[6];
  const int*   s_deg_l   = (const int*)d_in[7];
  const int*   s_r1      = (const int*)d_in[8];
  const int*   s_deg_r   = (const int*)d_in[9];
  const float* emb       = (const float*)d_in[10];
  const float* gcn_W     = (const float*)d_in[11];
  const float* gcn_lin_b = (const float*)d_in[12];
  const float* gcn_b     = (const float*)d_in[13];
  const float* gate_temp = (const float*)d_in[14];
  const float* cg_W1     = (const float*)d_in[15];
  const float* cg_b1     = (const float*)d_in[16];
  const float* cg_ln_g   = (const float*)d_in[17];
  const float* cg_ln_b   = (const float*)d_in[18];
  const float* cg_W2     = (const float*)d_in[19];
  const float* cg_b2     = (const float*)d_in[20];
  const float* se_W1     = (const float*)d_in[21];
  const float* se_b1     = (const float*)d_in[22];
  const float* se_W2     = (const float*)d_in[23];
  const float* se_b2     = (const float*)d_in[24];
  const float* se_ln_g   = (const float*)d_in[25];
  const float* se_ln_b   = (const float*)d_in[26];
  const float* lstm_Wih  = (const float*)d_in[27];
  const float* lstm_Whh  = (const float*)d_in[28];
  const float* lstm_bih  = (const float*)d_in[29];
  const float* lstm_bhh  = (const float*)d_in[30];

  char* base = (char*)d_ws;
  size_t off = 0;
  auto alloc = [&](size_t bytes) -> char* {
    off = (off + 255) & ~(size_t)255;
    char* p = base + off;
    off += bytes;
    return p;
  };
  const int B = 4096;
  unsigned short* wp_gcn  = (unsigned short*)alloc(256 * 128 * 2);
  unsigned short* wp_se1  = (unsigned short*)alloc(256 * 512 * 2);
  unsigned short* wp_se2  = (unsigned short*)alloc(512 * 256 * 2);
  unsigned short* wp_ih   = (unsigned short*)alloc(256 * 2048 * 2);
  unsigned short* wp_hh   = (unsigned short*)alloc(256 * 2048 * 2);
  float*          qvec    = (float*)alloc((size_t)B * 256 * 4);
  unsigned short* qvec_bf = (unsigned short*)alloc((size_t)B * 256 * 2);
  float*          svec    = (float*)alloc(16 * 256 * 4);
  unsigned short* svec_bf = (unsigned short*)alloc(16 * 256 * 2);
  unsigned short* h1q_bf  = (unsigned short*)alloc((size_t)B * 512 * 2);
  unsigned short* h1s_bf  = (unsigned short*)alloc(16 * 512 * 2);
  float*          qenc    = (float*)alloc((size_t)B * 256 * 4);
  unsigned short* qenc_bf = (unsigned short*)alloc((size_t)B * 256 * 2);
  float*          senc    = (float*)alloc(16 * 256 * 4);
  unsigned short* senc_bf = (unsigned short*)alloc(16 * 256 * 2);
  float*          sg      = (float*)alloc(256 * 4);
  float*          sgterm  = (float*)alloc(2048 * 4);
  float*          g0      = (float*)alloc((size_t)B * 2048 * 4);
  float*          gates   = (float*)alloc((size_t)B * 2048 * 4);
  float*          cbuf    = (float*)alloc((size_t)B * 512 * 4);
  float*          hbuf    = (float*)alloc((size_t)B * 256 * 4);
  unsigned short* hbf     = (unsigned short*)alloc((size_t)B * 256 * 2);

  // weight packing (bf16 WMMA B-fragment layouts)
  pack_b_kernel<<<8,   256, 0, stream>>>(gcn_W,    wp_gcn, 128,  256, 256);
  pack_b_kernel<<<32,  256, 0, stream>>>(se_W1,    wp_se1, 512,  256, 256);
  pack_b_kernel<<<32,  256, 0, stream>>>(se_W2,    wp_se2, 256,  512, 512);
  pack_b_kernel<<<128, 256, 0, stream>>>(lstm_Wih, wp_ih,  2048, 256, 256);
  pack_b_kernel<<<128, 256, 0, stream>>>(lstm_Whh, wp_hh,  2048, 256, 512);
  zpad_kernel<<<1, 256, 0, stream>>>(svec, svec_bf);

  // GCN neighbor encoders (query L/R, support L/R)
  neigh_kernel<<<B, 128, NB_SMEM, stream>>>(q_l1, q_deg_l, query,     emb, wp_gcn,
      gcn_lin_b, gcn_b, gate_temp, cg_W1, cg_b1, cg_ln_g, cg_ln_b, cg_W2, cg_b2,
      qvec, qvec_bf, 0);
  neigh_kernel<<<B, 128, NB_SMEM, stream>>>(q_r1, q_deg_r, query + 1, emb, wp_gcn,
      gcn_lin_b, gcn_b, gate_temp, cg_W1, cg_b1, cg_ln_g, cg_ln_b, cg_W2, cg_b2,
      qvec, qvec_bf, 128);
  neigh_kernel<<<5, 128, NB_SMEM, stream>>>(s_l1, s_deg_l, support,     emb, wp_gcn,
      gcn_lin_b, gcn_b, gate_temp, cg_W1, cg_b1, cg_ln_g, cg_ln_b, cg_W2, cg_b2,
      svec, svec_bf, 0);
  neigh_kernel<<<5, 128, NB_SMEM, stream>>>(s_r1, s_deg_r, support + 1, emb, wp_gcn,
      gcn_lin_b, gcn_b, gate_temp, cg_W1, cg_b1, cg_ln_g, cg_ln_b, cg_W2, cg_b2,
      svec, svec_bf, 128);

  // support_enc FFN (residual + LN), queries and supports
  gemm_relu512_kernel<<<B / 16, 256, 0, stream>>>(qvec_bf, wp_se1, se_b1, h1q_bf);
  gemm_relu512_kernel<<<1,      256, 0, stream>>>(svec_bf, wp_se1, se_b1, h1s_bf);
  gemm_ln256_kernel<<<B / 16, 256, 0, stream>>>(h1q_bf, wp_se2, se_b2, qvec,
                                                se_ln_g, se_ln_b, qenc, qenc_bf);
  gemm_ln256_kernel<<<1,      256, 0, stream>>>(h1s_bf, wp_se2, se_b2, svec,
                                                se_ln_g, se_ln_b, senc, senc_bf);

  support_mean_kernel<<<1, 256, 0, stream>>>(senc, sg);
  sgterm_kernel<<<8, 256, 0, stream>>>(lstm_Whh, sg, sgterm);

  // G0 = qenc @ Wih^T + (bih + bhh) ; constant across steps
  {
    dim3 grid(B / 64, 16);
    gemm_nt2048_kernel<<<grid, 128, 0, stream>>>(qenc_bf, wp_ih, nullptr,
                                                 lstm_bih, lstm_bhh, g0);
  }
  // step 0: gates = G0 (h_r = 0)
  lstm_cell_kernel<<<(B * 512) / 256, 256, 0, stream>>>(g0, cbuf, qenc, hbuf, hbf, 0);
  // steps 1..3: gates = G0 + h @ Whh_left^T + sgterm
  for (int step = 1; step < 4; ++step) {
    dim3 grid(B / 64, 16);
    gemm_nt2048_kernel<<<grid, 128, 0, stream>>>(hbf, wp_hh, g0, sgterm, nullptr, gates);
    lstm_cell_kernel<<<(B * 512) / 256, 256, 0, stream>>>(gates, cbuf, qenc, hbuf, hbf, step);
  }

  final_dot_kernel<<<B, 128, 0, stream>>>(hbuf, sg, (float*)d_out);
}
